// MultiHeadAttention_58059367907430
// MI455X (gfx1250) — compile-verified, matrix operand order checked by reading, not run on hardware
//
#include <hip/hip_runtime.h>

// ---------------------------------------------------------------------------
// MI455X (gfx1250) multi-head attention, bf16 WMMA pipeline, fp32 accumulate.
// GEMMs stage tiles with the Tensor Data Mover (TENSORcnt) double-buffered in
// LDS (async-to-LDS fallback); attention streams K/V fragments from L2 into
// WMMA with prefetch.
// ---------------------------------------------------------------------------

typedef __attribute__((ext_vector_type(16))) __bf16 v16bf;
typedef __attribute__((ext_vector_type(8)))  float  v8f;
typedef __attribute__((ext_vector_type(4)))  unsigned int tdm_v4u;
typedef __attribute__((ext_vector_type(8)))  int          tdm_v8i;
typedef __attribute__((ext_vector_type(4)))  int          tdm_v4i;

#define D_MODEL 1024
#define NHEADS  16
#define DKH     64
#define BATCH   2
#define SEQ     2048
#define MTOT    (BATCH * SEQ)   // 4096 rows

#if defined(__HIP_DEVICE_COMPILE__) && __has_builtin(__builtin_amdgcn_tensor_load_to_lds)
#define HAVE_TDM 1
#else
#define HAVE_TDM 0
#endif

// ---------------------------------------------------------------------------
// Fragment load: 16-bit A/B matrix 16x32 layout (CDNA5 ISA 7.12.2).
// lane 0-15  : row m = lane,    K = {0..7, 16..23}
// lane 16-31 : row m = lane-16, K = {8..15, 24..31}
// Works for both A (rows = M) and B (rows = N, since B is stored N x K).
// ---------------------------------------------------------------------------
__device__ __forceinline__ v16bf load_frag(const __bf16* __restrict__ base,
                                           int ld, int lane) {
    const int half = (lane >> 4) & 1;
    const int r    = lane & 15;
    const __bf16* p = base + (size_t)r * ld + half * 8;
    union { v16bf v; uint4 q[2]; } u;
    u.q[0] = *(const uint4*)(p);        // K = kb .. kb+7
    u.q[1] = *(const uint4*)(p + 16);   // K = kb+16 .. kb+23
    return u.v;
}

__device__ __forceinline__ v8f wmma_bf16(v16bf a, v16bf b, v8f c) {
    return __builtin_amdgcn_wmma_f32_16x16x32_bf16(
        /*neg_a=*/false, a, /*neg_b=*/false, b,
        /*c_mod=*/(short)0, c, /*reuse_a=*/false, /*reuse_b=*/false);
}

// Async global -> LDS, 16 bytes per lane, tracked by ASYNCcnt (fallback path).
__device__ __forceinline__ void async_g2l_b128(unsigned lds_addr,
                                               const void* gptr) {
    asm volatile("global_load_async_to_lds_b128 %0, %1, off"
                 :: "v"(lds_addr), "v"((unsigned long long)(uintptr_t)gptr)
                 : "memory");
}

#if HAVE_TDM
// ---------------------------------------------------------------------------
// TDM: move one rows x 32-bf16 tile of a row-major (rows_total x ld) bf16
// tensor into LDS with a padded row stride of 40 elements (80 B).
// D# per CDNA5 ISA 8.3/8.4:
//   pad_interval = 3 -> pad after every 16 DWORDs (64 B = one 32-elem row)
//   pad_amount   = 3 -> insert 4 DWORDs (16 B)  => LDS row stride 80 B
// ---------------------------------------------------------------------------
__device__ __forceinline__ void tdm_load_tile_32w(const __bf16* gtile,
                                                  unsigned lds_addr,
                                                  int ld, int rows) {
    unsigned long long ga = (unsigned long long)(uintptr_t)gtile;
    tdm_v4u g0;
    g0[0] = 1u;                                      // count=1, user mode
    g0[1] = lds_addr;                                // LDS byte address
    g0[2] = (unsigned)(ga & 0xFFFFFFFFull);          // global_addr[31:0]
    g0[3] = (unsigned)((ga >> 32) & 0x01FFFFFFull)   // global_addr[56:32]
          | 0x80000000u;                             // type=2 ("image")
    tdm_v8i g1;
    g1[0] = (int)((1u << 16)      // data_size = 1 -> 2 bytes
                | (1u << 20)      // pad_enable
                | (3u << 22)      // pad_interval: 16 DWORDs
                | (3u << 25));    // pad_amount:   4 DWORDs
    g1[1] = (int)(((unsigned)ld & 0xFFFFu) << 16);   // tensor_dim0[15:0]
    g1[2] = (int)(((unsigned)rows & 0xFFFFu) << 16); // tensor_dim1[15:0]
    g1[3] = (int)(32u << 16);                        // tile_dim0 = 32
    g1[4] = rows;                                    // tile_dim1 = rows
    g1[5] = ld;                                      // tensor_dim0_stride[31:0]
    g1[6] = 0;
    g1[7] = 0;
    tdm_v4i gz4 = {0, 0, 0, 0};
    tdm_v8i gz8 = {0, 0, 0, 0, 0, 0, 0, 0};
    // 6-arg form (clang-23 / therock headers): (g0, g1, g2, g3, g4, cpol)
    __builtin_amdgcn_tensor_load_to_lds(g0, g1, gz4, gz4, gz8, 0);
}
#endif

// ---------------------------------------------------------------------------
// fp32 -> bf16 cast
// ---------------------------------------------------------------------------
__global__ void cast_f32_bf16(const float* __restrict__ in,
                              __bf16* __restrict__ out, size_t n) {
    size_t i = (size_t)blockIdx.x * blockDim.x + threadIdx.x;
    size_t stride = (size_t)gridDim.x * blockDim.x;
    for (; i < n; i += stride) out[i] = (__bf16)in[i];
}

// ---------------------------------------------------------------------------
// GEMM: C[M,N] = A[M,K] * B[N,K]^T  (B row-major N x K == nn.Linear weight)
// 256 threads = 8 waves; tile 128x128; waves 4 (M) x 2 (N); k-step 32.
// Double-buffered LDS filled by TDM (TENSOR_LOAD_TO_LDS), one wave driving.
// OMODE 1: bf16 out, head-split (B,H,S,DK)             (Q, K)
// OMODE 2: bf16 out, head-split transposed (B,H,DK,S)  (V^T)
// OMODE 3: fp32 out, plain row-major M x N             (final projection)
// ---------------------------------------------------------------------------
#define TM 128
#define TN 128
#define TK 32
#define LDSS 40   // padded row stride (elements); 40*2 = 80 B (16B aligned)

template <int OMODE>
__global__ __launch_bounds__(256)
void gemm_bf16_wmma(const __bf16* __restrict__ A, const __bf16* __restrict__ B,
                    void* __restrict__ out, int M, int N, int K) {
    __shared__ __attribute__((aligned(16))) __bf16 As[2][TM * LDSS];
    __shared__ __attribute__((aligned(16))) __bf16 Bs[2][TN * LDSS];

    const int tid    = threadIdx.x;
    const int lane   = tid & 31;
    const int wid    = tid >> 5;
    const int wave_m = wid & 3;   // 0..3 -> 32 rows each
    const int wave_n = wid >> 2;  // 0..1 -> 64 cols each
    const int bm     = blockIdx.y * TM;
    const int bn     = blockIdx.x * TN;

    v8f acc[2][4];
    const v8f vz = {0.f,0.f,0.f,0.f,0.f,0.f,0.f,0.f};
#pragma unroll
    for (int i = 0; i < 2; ++i)
#pragma unroll
        for (int j = 0; j < 4; ++j) acc[i][j] = vz;

#if HAVE_TDM
    // One wave drives the Tensor Data Mover: one descriptor per 128x32 tile.
    const bool issuer = (wid == 0);
    auto issue_tile = [&](int buf, int k0) {
        if (issuer) {
            tdm_load_tile_32w(A + (size_t)bm * K + k0,
                              (unsigned)(uintptr_t)&As[buf][0], K, TM);
            tdm_load_tile_32w(B + (size_t)bn * K + k0,
                              (unsigned)(uintptr_t)&Bs[buf][0], K, TN);
        }
    };
#else
    // Fallback: async global->LDS, 512 16B chunks per tile, 2 per thread.
    auto issue_tile = [&](int buf, int k0) {
#pragma unroll
        for (int j = 0; j < 2; ++j) {
            int chunk = tid + j * 256;
            int row   = chunk >> 2;
            int c16   = chunk & 3;
            unsigned la = (unsigned)(uintptr_t)&As[buf][row * LDSS + c16 * 8];
            async_g2l_b128(la, A + (size_t)(bm + row) * K + k0 + c16 * 8);
            unsigned lb = (unsigned)(uintptr_t)&Bs[buf][row * LDSS + c16 * 8];
            async_g2l_b128(lb, B + (size_t)(bn + row) * K + k0 + c16 * 8);
        }
    };
#endif

    issue_tile(0, 0);
    const int nk = K / TK;
    for (int kt = 0; kt < nk; ++kt) {
        const int p = kt & 1;
        // TDM/async ops complete in order per wave: after issuing the next
        // tile's transfers, "<= just-issued outstanding" means the current
        // tile has fully landed in LDS.
#if HAVE_TDM
        if (kt + 1 < nk) {
            issue_tile(1 - p, (kt + 1) * TK);
            if (issuer) __builtin_amdgcn_s_wait_tensorcnt((short)2);
        } else {
            if (issuer) __builtin_amdgcn_s_wait_tensorcnt((short)0);
        }
#else
        if (kt + 1 < nk) {
            issue_tile(1 - p, (kt + 1) * TK);
            asm volatile("s_wait_asynccnt 0x4" ::: "memory");
        } else {
            asm volatile("s_wait_asynccnt 0x0" ::: "memory");
        }
#endif
        __syncthreads();

        v16bf af[2], bf[4];
#pragma unroll
        for (int im = 0; im < 2; ++im)
            af[im] = load_frag(&As[p][(wave_m * 32 + im * 16) * LDSS], LDSS, lane);
#pragma unroll
        for (int jn = 0; jn < 4; ++jn)
            bf[jn] = load_frag(&Bs[p][(wave_n * 64 + jn * 16) * LDSS], LDSS, lane);

#pragma unroll
        for (int im = 0; im < 2; ++im)
#pragma unroll
            for (int jn = 0; jn < 4; ++jn)
                acc[im][jn] = wmma_bf16(af[im], bf[jn], acc[im][jn]);
        __syncthreads();   // reads done before next transfer overwrites buffer
    }

    // store: C layout -> row = i + 8*half, col = lane & 15
    const int half = (lane >> 4) & 1;
    const int nl   = lane & 15;
#pragma unroll
    for (int im = 0; im < 2; ++im) {
#pragma unroll
        for (int jn = 0; jn < 4; ++jn) {
#pragma unroll
            for (int i = 0; i < 8; ++i) {
                int m = bm + wave_m * 32 + im * 16 + i + half * 8;
                int n = bn + wave_n * 64 + jn * 16 + nl;
                float v = acc[im][jn][i];
                if (OMODE == 3) {
                    ((float*)out)[(size_t)m * N + n] = v;
                } else {
                    int b = m >> 11;          // m / SEQ
                    int s = m & (SEQ - 1);
                    int h = n >> 6;           // n / DKH
                    int d = n & (DKH - 1);
                    if (OMODE == 1) {
                        ((__bf16*)out)[(((size_t)(b * NHEADS + h) * SEQ + s) << 6) + d] =
                            (__bf16)v;
                    } else { // OMODE == 2: transposed V -> (B,H,DK,S)
                        ((__bf16*)out)[((size_t)(b * NHEADS + h) * DKH + d) * SEQ + s] =
                            (__bf16)v;
                    }
                }
            }
        }
    }
}

// ---------------------------------------------------------------------------
// Causal flash attention.  Block = 128 threads = 4 waves; each wave owns 16
// consecutive query rows; block covers 64 q-rows of one (b,h).
// Q,K: (B,H,S,DK) bf16.  Vt: (B,H,DK,S) bf16.  Out: (B,S,D_MODEL) bf16.
// ---------------------------------------------------------------------------
#define PST 72    // P LDS row stride (elements); 72*2 = 144 B (16B aligned)

__global__ __launch_bounds__(128)
void flash_attn_wmma(const __bf16* __restrict__ Q, const __bf16* __restrict__ Kb,
                     const __bf16* __restrict__ Vt, __bf16* __restrict__ outA) {
    __shared__ __attribute__((aligned(16))) __bf16 Plds[4 * 16 * PST];

    const int tid  = threadIdx.x;
    const int lane = tid & 31;
    const int wid  = tid >> 5;
    const int half = (lane >> 4) & 1;
    const int nl   = lane & 15;

    const int nq64 = SEQ >> 6;                 // 32 q-chunks per head
    int idx = blockIdx.x;
    int qc  = idx % nq64;
    int h   = (idx / nq64) % NHEADS;
    int b   = idx / (nq64 * NHEADS);
    const int qbase = qc * 64 + wid * 16;

    const size_t head = (size_t)(b * NHEADS + h) * SEQ * DKH;
    const __bf16* Qh = Q  + head;
    const __bf16* Kh = Kb + head;
    const __bf16* Vh = Vt + head;              // (DK, S) per head
    __bf16* pl = Plds + wid * 16 * PST;

    // Q fragments for this wave's 16 rows, two k-steps over d_k = 64
    v16bf qf[2];
    qf[0] = load_frag(Qh + (size_t)qbase * DKH,      DKH, lane);
    qf[1] = load_frag(Qh + (size_t)qbase * DKH + 32, DKH, lane);

    const v8f vz = {0.f,0.f,0.f,0.f,0.f,0.f,0.f,0.f};
    v8f o[4];
#pragma unroll
    for (int t = 0; t < 4; ++t) o[t] = vz;

    float mrow[8], lrow[8];
#pragma unroll
    for (int i = 0; i < 8; ++i) { mrow[i] = -__builtin_inff(); lrow[i] = 0.f; }

    const float scale = 0.125f;  // 1/sqrt(64)

    for (int kv = 0; kv <= qbase + 15; kv += 64) {
        // prefetch next key/value tiles (speculative; dropped if OOB)
        __builtin_prefetch(Kh + (size_t)(kv + 64) * DKH, 0, 1);
        __builtin_prefetch(Vh + kv + 64, 0, 1);

        // ---- scores S = Q K^T : 4 n-tiles x 2 k-steps -------------------
        v8f sc[4];
#pragma unroll
        for (int nt = 0; nt < 4; ++nt) sc[nt] = vz;
#pragma unroll
        for (int nt = 0; nt < 4; ++nt)
#pragma unroll
            for (int ks = 0; ks < 2; ++ks) {
                v16bf kf = load_frag(Kh + (size_t)(kv + nt * 16) * DKH + ks * 32,
                                     DKH, lane);
                sc[nt] = wmma_bf16(qf[ks], kf, sc[nt]);
            }

        // ---- scale + causal mask ---------------------------------------
#pragma unroll
        for (int nt = 0; nt < 4; ++nt)
#pragma unroll
            for (int i = 0; i < 8; ++i) {
                int qi = qbase + i + half * 8;
                int kj = kv + nt * 16 + nl;
                float v = sc[nt][i] * scale;
                sc[nt][i] = (kj > qi) ? -__builtin_inff() : v;
            }

        // ---- row max (across 16 lanes of each half-wave) ---------------
        float rmax[8];
#pragma unroll
        for (int i = 0; i < 8; ++i) {
            float v = sc[0][i];
#pragma unroll
            for (int nt = 1; nt < 4; ++nt) v = fmaxf(v, sc[nt][i]);
#pragma unroll
            for (int d = 1; d < 16; d <<= 1)
                v = fmaxf(v, __shfl_xor(v, d, 32));
            rmax[i] = v;
        }

        float mnew[8], cf[8];
#pragma unroll
        for (int i = 0; i < 8; ++i) {
            mnew[i] = fmaxf(mrow[i], rmax[i]);
            cf[i]   = __expf(mrow[i] - mnew[i]);
            mrow[i] = mnew[i];
        }

        // ---- P = exp(S - m), row sums ----------------------------------
        float rsum[8];
#pragma unroll
        for (int i = 0; i < 8; ++i) rsum[i] = 0.f;
#pragma unroll
        for (int nt = 0; nt < 4; ++nt)
#pragma unroll
            for (int i = 0; i < 8; ++i) {
                float p = __expf(sc[nt][i] - mnew[i]);
                sc[nt][i] = p;
                rsum[i] += p;
            }
#pragma unroll
        for (int i = 0; i < 8; ++i) {
#pragma unroll
            for (int d = 1; d < 16; d <<= 1)
                rsum[i] += __shfl_xor(rsum[i], d, 32);
            lrow[i] = lrow[i] * cf[i] + rsum[i];
        }

        // ---- rescale O, stash P to LDS (C-layout -> A-frag layout) -----
#pragma unroll
        for (int t = 0; t < 4; ++t)
#pragma unroll
            for (int i = 0; i < 8; ++i) o[t][i] *= cf[i];

#pragma unroll
        for (int nt = 0; nt < 4; ++nt)
#pragma unroll
            for (int i = 0; i < 8; ++i)
                pl[(i + half * 8) * PST + nt * 16 + nl] = (__bf16)sc[nt][i];

        asm volatile("s_wait_dscnt 0x0" ::: "memory");  // per-wave LDS RAW

        // ---- O += P V : 4 d-tiles x 2 k-steps over the 64 keys ---------
        v16bf pf[2];
        pf[0] = load_frag(pl,      PST, lane);
        pf[1] = load_frag(pl + 32, PST, lane);
#pragma unroll
        for (int dt = 0; dt < 4; ++dt)
#pragma unroll
            for (int ks = 0; ks < 2; ++ks) {
                v16bf vf = load_frag(Vh + (size_t)(dt * 16) * SEQ + kv + ks * 32,
                                     SEQ, lane);
                o[dt] = wmma_bf16(pf[ks], vf, o[dt]);
            }
    }

    // ---- normalize and store merged-head output ------------------------
    float inv[8];
#pragma unroll
    for (int i = 0; i < 8; ++i) inv[i] = 1.0f / lrow[i];
#pragma unroll
    for (int dt = 0; dt < 4; ++dt)
#pragma unroll
        for (int i = 0; i < 8; ++i) {
            int qi = qbase + i + half * 8;
            int d  = dt * 16 + nl;
            outA[((size_t)b * SEQ + qi) * D_MODEL + h * DKH + d] =
                (__bf16)(o[dt][i] * inv[i]);
        }
}

// ---------------------------------------------------------------------------
// Host launcher
// ---------------------------------------------------------------------------
extern "C" void kernel_launch(void* const* d_in, const int* in_sizes, int n_in,
                              void* d_out, int out_size, void* d_ws, size_t ws_size,
                              hipStream_t stream) {
    const float* x  = (const float*)d_in[0];
    const float* Wq = (const float*)d_in[1];
    const float* Wk = (const float*)d_in[2];
    const float* Wv = (const float*)d_in[3];
    const float* Wo = (const float*)d_in[4];

    constexpr size_t NX = (size_t)MTOT * D_MODEL;        // 4,194,304
    constexpr size_t NW = (size_t)D_MODEL * D_MODEL;     // 1,048,576

    char* ws = (char*)d_ws;
    __bf16* xb   = (__bf16*)(ws);                         //  8 MB
    __bf16* wqb  = (__bf16*)(ws + (8u << 20));            //  2 MB
    __bf16* wkb  = (__bf16*)(ws + (10u << 20));           //  2 MB
    __bf16* wvb  = (__bf16*)(ws + (12u << 20));           //  2 MB
    __bf16* wob  = (__bf16*)(ws + (14u << 20));           //  2 MB
    __bf16* Qb   = (__bf16*)(ws + (16u << 20));           //  8 MB
    __bf16* Kb   = (__bf16*)(ws + (24u << 20));           //  8 MB
    __bf16* Vtb  = (__bf16*)(ws + (32u << 20));           //  8 MB
    __bf16* attb = (__bf16*)(ws + (40u << 20));           //  8 MB

    cast_f32_bf16<<<1024, 256, 0, stream>>>(x,  xb,  NX);
    cast_f32_bf16<<<512,  256, 0, stream>>>(Wq, wqb, NW);
    cast_f32_bf16<<<512,  256, 0, stream>>>(Wk, wkb, NW);
    cast_f32_bf16<<<512,  256, 0, stream>>>(Wv, wvb, NW);
    cast_f32_bf16<<<512,  256, 0, stream>>>(Wo, wob, NW);

    dim3 gg(D_MODEL / TN, MTOT / TM);  // (8, 32)
    gemm_bf16_wmma<1><<<gg, 256, 0, stream>>>(xb, wqb, Qb,  MTOT, D_MODEL, D_MODEL);
    gemm_bf16_wmma<1><<<gg, 256, 0, stream>>>(xb, wkb, Kb,  MTOT, D_MODEL, D_MODEL);
    gemm_bf16_wmma<2><<<gg, 256, 0, stream>>>(xb, wvb, Vtb, MTOT, D_MODEL, D_MODEL);

    flash_attn_wmma<<<BATCH * NHEADS * (SEQ / 64), 128, 0, stream>>>(Qb, Kb, Vtb, attb);

    gemm_bf16_wmma<3><<<gg, 256, 0, stream>>>(attb, wob, d_out, MTOT, D_MODEL, D_MODEL);
}
